// BahdanauAttention_23613730194039
// MI455X (gfx1250) — compile-verified, matrix operand order checked by reading, not run on hardware
//
#include <hip/hip_runtime.h>
#include <hip/hip_bf16.h>
#include <math.h>

typedef __bf16 bf16_t;
typedef __attribute__((ext_vector_type(16))) __bf16 v16bf;
typedef __attribute__((ext_vector_type(8)))  float  v8f;

#define Bsz 64
#define Ssz 2048
#define Hsz 512

// ---------------------------------------------------------------------------
// Kernel 1: W_h (K=H x N=H, f32, row-major) -> Wt (N-major, bf16)
// Makes each 16-column WMMA B-slab a contiguous 16KB region.
// ---------------------------------------------------------------------------
__global__ void __launch_bounds__(256)
k_wt(const float* __restrict__ Wh, bf16_t* __restrict__ Wt) {
  int idx = blockIdx.x * 256 + threadIdx.x;      // 0 .. H*H-1
  int n = idx >> 9, k = idx & 511;
  Wt[n * Hsz + k] = (bf16_t)Wh[k * Hsz + n];
}

// ---------------------------------------------------------------------------
// Kernel 2: qs = query @ W_s   (64x512 @ 512x512 — tiny, FMA is fine)
// ---------------------------------------------------------------------------
__global__ void __launch_bounds__(512)
k_qs(const float* __restrict__ q, const float* __restrict__ Ws,
     float* __restrict__ qs) {
  int b = blockIdx.x, h = threadIdx.x;
  float acc = 0.f;
  for (int k = 0; k < Hsz; ++k)
    acc = fmaf(q[b * Hsz + k], Ws[k * Hsz + h], acc);
  qs[b * Hsz + h] = acc;
}

// ---------------------------------------------------------------------------
// Kernel 3 (WMMA): scores[b,s] = v . tanh(enc[b,s,:]@W_h + qs[b,:])
// Block = 64 threads (2 waves), covers one b and 32 s-rows.
//  - encoder tile (32x512) staged once to LDS as bf16  (single HBM pass)
//  - per N-tile: 16x512 bf16 W-slab staged to LDS, 16 wmma ops over K=512
// ---------------------------------------------------------------------------
__global__ void __launch_bounds__(64)
k_scores(const float* __restrict__ enc, const bf16_t* __restrict__ Wt,
         const float* __restrict__ qs, const float* __restrict__ vvec,
         float* __restrict__ scores) {
  __shared__ bf16_t sA[32 * Hsz];   // 32 KB: 32 s-rows x 512 K
  __shared__ bf16_t sB[16 * Hsz];   // 16 KB: 16 n-cols x 512 K (n-major)

  int b  = blockIdx.x >> 6;                 // 64 tiles of 32 rows per batch
  int s0 = (blockIdx.x & 63) * 32;
  int tid  = threadIdx.x;
  int lane = tid & 31;
  int wave = tid >> 5;

  // Stage encoder tile: 32 rows x 512 f32 -> bf16 LDS (float4 coalesced)
  const float4* enc4 = (const float4*)(enc + ((size_t)b * Ssz + s0) * Hsz);
  for (int i = tid; i < 32 * 128; i += 64) {
    int row = i >> 7, c4 = i & 127;
    float4 x = enc4[row * 128 + c4];
    bf16_t* dst = &sA[row * Hsz + c4 * 4];
    dst[0] = (bf16_t)x.x; dst[1] = (bf16_t)x.y;
    dst[2] = (bf16_t)x.z; dst[3] = (bf16_t)x.w;
  }

  float sacc[8];
  #pragma unroll
  for (int r = 0; r < 8; ++r) sacc[r] = 0.f;

  const int nsel  = lane & 15;          // M row (A) / N col (B) for this lane
  const int khalf = (lane >> 4) << 4;   // lanes 16-31 take the upper 16 K

  for (int nt = 0; nt < 32; ++nt) {
    __syncthreads();
    // Stage W-slab for this 16-column N-tile (contiguous 16 KB, L2-hot)
    const uint4* src  = (const uint4*)(Wt + (size_t)nt * 16 * Hsz);
    uint4*       dstB = (uint4*)sB;
    for (int i = tid; i < 1024; i += 64) dstB[i] = src[i];
    __syncthreads();

    v8f c = {};
    #pragma unroll
    for (int kk = 0; kk < 16; ++kk) {
      int k0 = kk * 32 + khalf;
      v16bf a  = *(const v16bf*)&sA[(wave * 16 + nsel) * Hsz + k0];
      v16bf bb = *(const v16bf*)&sB[nsel * Hsz + k0];
      c = __builtin_amdgcn_wmma_f32_16x16x32_bf16(
              false, a, false, bb, (short)0, c, false, false);
    }

    // Fold this N-tile into per-row score partials: tanh(c + qs[n]) * v[n]
    int   n  = nt * 16 + nsel;
    float qv = qs[b * Hsz + n];
    float vv = vvec[n];
    #pragma unroll
    for (int r = 0; r < 8; ++r)
      sacc[r] += tanhf(c[r] + qv) * vv;
  }

  // Reduce across the 16 lanes holding the 16 N columns of each half-wave.
  #pragma unroll
  for (int r = 0; r < 8; ++r) {
    float s = sacc[r];
    s += __shfl_xor(s, 1, 32);
    s += __shfl_xor(s, 2, 32);
    s += __shfl_xor(s, 4, 32);
    s += __shfl_xor(s, 8, 32);
    if ((lane & 15) == 0) {
      int row = r + ((lane >> 4) << 3);   // C layout: lanes 0-15 M=r, 16-31 M=8+r
      scores[(size_t)b * Ssz + s0 + wave * 16 + row] = s;
    }
  }
}

// ---------------------------------------------------------------------------
// Kernel 4: masked softmax over S per batch -> attn (written into d_out slot)
// ---------------------------------------------------------------------------
__global__ void __launch_bounds__(256)
k_softmax(const float* __restrict__ scores, const int* __restrict__ lens,
          float* __restrict__ attn) {
  __shared__ float red[256];
  int b = blockIdx.x, tid = threadIdx.x;
  int len = lens[b];

  float vals[8];
  float lmax = -INFINITY;
  #pragma unroll
  for (int i = 0; i < 8; ++i) {
    int s = tid + i * 256;
    float x = (s < len) ? scores[(size_t)b * Ssz + s] : -INFINITY;
    vals[i] = x;
    lmax = fmaxf(lmax, x);
  }
  red[tid] = lmax; __syncthreads();
  for (int off = 128; off > 0; off >>= 1) {
    if (tid < off) red[tid] = fmaxf(red[tid], red[tid + off]);
    __syncthreads();
  }
  float m = red[0]; __syncthreads();

  float lsum = 0.f;
  #pragma unroll
  for (int i = 0; i < 8; ++i) {
    float e = __expf(vals[i] - m);   // masked -> exp(-inf)=0
    vals[i] = e; lsum += e;
  }
  red[tid] = lsum; __syncthreads();
  for (int off = 128; off > 0; off >>= 1) {
    if (tid < off) red[tid] += red[tid + off];
    __syncthreads();
  }
  float inv = 1.f / red[0];
  #pragma unroll
  for (int i = 0; i < 8; ++i)
    attn[(size_t)b * Ssz + tid + i * 256] = vals[i] * inv;
}

// ---------------------------------------------------------------------------
// Kernel 5: context[b,h] = sum_s attn[b,s] * enc[b,s,h]  (2nd encoder pass)
// ---------------------------------------------------------------------------
__global__ void __launch_bounds__(256)
k_context(const float* __restrict__ enc, const float* __restrict__ attn,
          float* __restrict__ ctx) {
  int b = blockIdx.x >> 1;
  int h = ((blockIdx.x & 1) << 8) + threadIdx.x;
  const float* e = enc + (size_t)b * Ssz * Hsz + h;
  const float* a = attn + (size_t)b * Ssz;
  float acc = 0.f;
  for (int s = 0; s < Ssz; ++s) {
    __builtin_prefetch(&e[(size_t)(s + 8) * Hsz], 0, 0);
    acc = fmaf(a[s], e[(size_t)s * Hsz], acc);
  }
  ctx[b * Hsz + h] = acc;
}

// ---------------------------------------------------------------------------
// Kernel 6: w_out = tanh([context, query] @ W_out)
// ---------------------------------------------------------------------------
__global__ void __launch_bounds__(512)
k_out(const float* __restrict__ ctx, const float* __restrict__ q,
      const float* __restrict__ Wout, float* __restrict__ wout) {
  int b = blockIdx.x, h = threadIdx.x;
  float acc = 0.f;
  for (int k = 0; k < Hsz; ++k)
    acc = fmaf(ctx[b * Hsz + k], Wout[k * Hsz + h], acc);
  for (int k = 0; k < Hsz; ++k)
    acc = fmaf(q[b * Hsz + k], Wout[(Hsz + k) * Hsz + h], acc);
  wout[b * Hsz + h] = tanhf(acc);
}

// ---------------------------------------------------------------------------
extern "C" void kernel_launch(void* const* d_in, const int* in_sizes, int n_in,
                              void* d_out, int out_size, void* d_ws, size_t ws_size,
                              hipStream_t stream) {
  const float* query = (const float*)d_in[0];
  const float* enc   = (const float*)d_in[1];
  const int*   lens  = (const int*)  d_in[2];
  const float* W_h   = (const float*)d_in[3];
  const float* W_s   = (const float*)d_in[4];
  const float* vvec  = (const float*)d_in[5];
  const float* W_out = (const float*)d_in[6];

  float* out   = (float*)d_out;
  float* w_out = out;                 // (B,1,H)  = 64*512
  float* attn  = out + Bsz * Hsz;     // (B,S,1)  = 64*2048

  // Workspace layout (all 16B aligned): Wt bf16 | qs | scores | context
  char*   ws      = (char*)d_ws;
  bf16_t* Wt      = (bf16_t*)ws;                               // 512 KB
  float*  qs      = (float*)(ws + 524288);                     // 128 KB
  float*  scores  = (float*)(ws + 524288 + 131072);            // 512 KB
  float*  context = (float*)(ws + 524288 + 131072 + 524288);   // 128 KB

  k_wt     <<<Hsz * Hsz / 256, 256, 0, stream>>>(W_h, Wt);
  k_qs     <<<Bsz, Hsz, 0, stream>>>(query, W_s, qs);
  k_scores <<<Bsz * (Ssz / 32), 64, 0, stream>>>(enc, Wt, qs, vvec, scores);
  k_softmax<<<Bsz, 256, 0, stream>>>(scores, lens, attn);
  k_context<<<Bsz * 2, 256, 0, stream>>>(enc, attn, context);
  k_out    <<<Bsz, Hsz, 0, stream>>>(context, query, W_out, w_out);
}
